// RssiFuseV2_88553635709445
// MI455X (gfx1250) — matrix operand order repaired and checked by reading, not compiled, not run on hardware
//
#include <hip/hip_runtime.h>
#include <math.h>

typedef __attribute__((ext_vector_type(16))) __bf16 v16bf;
typedef __attribute__((ext_vector_type(8)))  __bf16 v8bf;
typedef __attribute__((ext_vector_type(8)))  float  v8f;
typedef __attribute__((ext_vector_type(4)))  unsigned int v4u;
typedef __attribute__((ext_vector_type(8)))  int v8i;
typedef __attribute__((ext_vector_type(4)))  int v4i;

#define HW_PLANE 16384   // H*W
#define W_DIM    128
#define C_DIM    64
#define B_DIM    16
#define W_LD     68      // padded LDS row stride (dwords) for the weight tile
#define N_TOTAL  (B_DIM * C_DIM * HW_PLANE)   // 16,777,216 elems per tensor

static __device__ __forceinline__ int lane_id() { return threadIdx.x & 31; }

// ---- Tensor Data Mover: 2-D tile load, global -> LDS (D# per ISA 08 §8) ----
static __device__ __forceinline__ void tdm_load_2d(
    unsigned lds_byte_addr, const void* gptr,
    unsigned data_size_code,                    // 0=1B,1=2B,2=4B,3=8B
    unsigned tensor_d0, unsigned tensor_d1,     // tensor extents (elems)
    unsigned tile_d0, unsigned tile_d1,         // tile extents (elems)
    unsigned long long stride0,                 // dim0 stride (elems)
    unsigned pad_enable, unsigned pad_interval_code, unsigned pad_amount_code) {
  unsigned long long ga = (unsigned long long)(size_t)gptr;
  v4u g0;
  g0[0] = 1u;                                          // count=1, user descriptor
  g0[1] = lds_byte_addr;                               // lds_addr
  g0[2] = (unsigned)(ga & 0xffffffffu);                // global_addr[31:0]
  g0[3] = (unsigned)((ga >> 32) & 0x01ffffffu) | (2u << 30);  // addr[56:32] | type=2
  v8i g1;
  g1[0] = (int)((data_size_code << 16) | (pad_enable << 20) |
                (pad_interval_code << 22) | (pad_amount_code << 25));
  g1[1] = (int)((tensor_d0 & 0xffffu) << 16);          // tensor_dim0[15:0] @ bits63:48
  g1[2] = (int)(((tensor_d0 >> 16) & 0xffffu) |        // tensor_dim0[31:16]
                ((tensor_d1 & 0xffffu) << 16));        // tensor_dim1[15:0]
  g1[3] = (int)(((tensor_d1 >> 16) & 0xffffu) |        // tensor_dim1[31:16]
                ((tile_d0 & 0xffffu) << 16));          // tile_dim0
  g1[4] = (int)(tile_d1 & 0xffffu);                    // tile_dim1 (tile_dim2 = 0)
  g1[5] = (int)(stride0 & 0xffffffffu);                // tensor_dim0_stride[31:0]
  g1[6] = (int)((stride0 >> 32) & 0xffffu);            // stride[47:32] (dim1_stride=0)
  g1[7] = 0;
  v4i z4 = {0, 0, 0, 0};
#if defined(__clang_major__) && __clang_major__ >= 23
  v8i z8 = {0, 0, 0, 0, 0, 0, 0, 0};
  __builtin_amdgcn_tensor_load_to_lds(g0, g1, z4, z4, z8, 0);
#else
  __builtin_amdgcn_tensor_load_to_lds(g0, g1, z4, z4, 0);
#endif
}

// ---- WMMA fragment loaders (contiguous k order k = 16*half + s, same in A and B) -------------------
// A fragment 16x32 (bf16 row-major source, leading dim ld):
//   lane<16 : row=row0+lane,     K = k0+{0..15}
//   lane>=16: row=row0+(lane-16),K = k0+{16..31}
static __device__ __forceinline__ v16bf load_a_bf16(const __bf16* base, int row0, int k0, int ld) {
  int lane = lane_id(), half = lane >> 4;
  const __bf16* p = base + (size_t)(row0 + (lane & 15)) * ld + k0 + half * 16;
  v8bf lo = *(const v8bf*)p;
  v8bf hi = *(const v8bf*)(p + 8);
  v16bf f;
#pragma unroll
  for (int i = 0; i < 8; ++i) { f[i] = lo[i]; f[8 + i] = hi[i]; }
  return f;
}

// Same A layout, converting from an f32 row-major source (LDS weight tile)
static __device__ __forceinline__ v16bf load_a_f32(const float* base, int row0, int k0, int ld) {
  int lane = lane_id(), half = lane >> 4;
  const float* p = base + (size_t)(row0 + (lane & 15)) * ld + k0 + half * 16;
  v16bf f;
#pragma unroll
  for (int i = 0; i < 8; ++i) { f[i] = (__bf16)p[i]; f[8 + i] = (__bf16)p[8 + i]; }
  return f;
}

// B fragment 32x16 where B = M^T (M row-major bf16, ld): B[k,n] = M[col0+n, k]
static __device__ __forceinline__ v16bf load_bT_bf16(const __bf16* M, int col0, int k0, int ld) {
  int lane = lane_id(), half = lane >> 4;
  const __bf16* p = M + (size_t)(col0 + (lane & 15)) * ld + k0 + half * 16;
  return *(const v16bf*)p;
}

// B fragment 32x16 from f32 row-major X (LDS tile): B[k,n] = X[k0+k, n0+n]
static __device__ __forceinline__ v16bf load_b_col_f32(const float* X, int k0, int n0, int ld) {
  int lane = lane_id(), half = lane >> 4;
  const float* p = X + (size_t)(k0 + half * 16) * ld + (n0 + (lane & 15));
  v16bf f;
#pragma unroll
  for (int i = 0; i < 16; ++i) f[i] = (__bf16)p[(size_t)i * ld];
  return f;
}

static __device__ __forceinline__ v8f wmma_bf16(v16bf a, v16bf b, v8f c) {
  return __builtin_amdgcn_wmma_f32_16x16x32_bf16(false, a, false, b, (short)0, c, false, false);
}

// ---- Kernel 0: zero per-channel stats --------------------------------------
__global__ void RF_init_stats(float* stats) {
  if (threadIdx.x < 2 * C_DIM) stats[threadIdx.x] = 0.f;
}

// ---- Kernel 1: 1x1-conv projections, f32 in -> bf16 out --------------------
// X tile (64x128) and W (64x64) are staged into LDS by the Tensor Data Mover;
// the W descriptor uses TDM LDS-padding (4 dwords per 64) so A-fragment reads
// don't alias a single LDS bank.  V outputs are stored H/W-transposed so the
// attention S*V GEMM gets contiguous B-fragment loads.
__global__ __launch_bounds__(256) void RF_proj(
    const float* __restrict__ x1, const float* __restrict__ x2,
    const float* __restrict__ rssi1, const float* __restrict__ rssi2,
    const float* __restrict__ qw, const float* __restrict__ qb,
    const float* __restrict__ kw, const float* __restrict__ kb,
    const float* __restrict__ vw, const float* __restrict__ vb,
    __bf16* Q1, __bf16* K1, __bf16* V1T, __bf16* Q2, __bf16* K2, __bf16* V2T) {
  __shared__ float xs[C_DIM][W_DIM];   // 32 KB input tile
  __shared__ float wsm[C_DIM][W_LD];   // 17 KB padded weight tile
  const float *Wm, *bias, *X; __bf16* out; bool transp = false;
  switch (blockIdx.z) {
    case 0:  Wm = qw; bias = qb; X = rssi1; out = Q1; break;
    case 1:  Wm = kw; bias = kb; X = x1;    out = K1; break;
    case 2:  Wm = vw; bias = vb; X = x1;    out = V1T; transp = true; break;
    case 3:  Wm = qw; bias = qb; X = rssi2; out = Q2; break;
    case 4:  Wm = kw; bias = kb; X = x2;    out = K2; break;
    default: Wm = vw; bias = vb; X = x2;    out = V2T; transp = true; break;
  }
  int b  = blockIdx.y;
  int n0 = blockIdx.x * 128;
  const float* Xb = X + (size_t)b * C_DIM * HW_PLANE + n0;   // tile start

  if (threadIdx.x < 32) {              // one wave issues both TDM descriptors
    tdm_load_2d((unsigned)(size_t)(void*)&xs[0][0], Xb, 2,
                HW_PLANE, C_DIM, W_DIM, C_DIM, HW_PLANE, 0, 0, 0);
    tdm_load_2d((unsigned)(size_t)(void*)&wsm[0][0], Wm, 2,
                C_DIM, C_DIM, C_DIM, C_DIM, C_DIM,
                1, 5, 3);              // pad 4 dwords every 64 dwords -> ld 68
    __builtin_amdgcn_s_wait_tensorcnt(0);
  }
  __syncthreads();

  int wave = threadIdx.x >> 5;
  int nloc = wave * 16;                // 8 waves x 16 cols
  v8f acc[4] = {};                     // 4 M-tiles of 16
#pragma unroll
  for (int kk = 0; kk < 2; ++kk) {     // K = 64 = 2 x 32
    int k0 = kk * 32;
    v16bf bf = load_b_col_f32(&xs[0][0], k0, nloc, W_DIM);
#pragma unroll
    for (int mt = 0; mt < 4; ++mt) {
      v16bf af = load_a_f32(&wsm[0][0], mt * 16, k0, W_LD);
      acc[mt] = wmma_bf16(af, bf, acc[mt]);
    }
  }
  int lane = lane_id(), half = lane >> 4;
  int n    = n0 + nloc + (lane & 15);
  int pos  = transp ? ((n & 127) * W_DIM + (n >> 7)) : n;   // H<->W transpose
#pragma unroll
  for (int mt = 0; mt < 4; ++mt) {
#pragma unroll
    for (int j = 0; j < 8; ++j) {
      int c = mt * 16 + j + 8 * half;  // C/D layout: M = j + 8*half
      float v = acc[mt][j] + bias[c];
      out[((size_t)b * C_DIM + c) * HW_PLANE + pos] = (__bf16)v;
    }
  }
}

// ---- Kernel 2: attention per (b,c) plane, both branches fused --------------
__global__ __launch_bounds__(256) void RF_attn(
    const __bf16* __restrict__ Q1, const __bf16* __restrict__ K1, const __bf16* __restrict__ V1T,
    const __bf16* __restrict__ Q2, const __bf16* __restrict__ K2, const __bf16* __restrict__ V2T,
    float* __restrict__ fused, float* chSum, float* chSumSq) {
  __shared__ __bf16 lds_s[8][16][W_DIM];             // per-wave softmax(S) tile
  int bc   = blockIdx.x;
  int c    = bc & (C_DIM - 1);
  int wave = threadIdx.x >> 5;
  int lane = lane_id(), half = lane >> 4;
  int h0   = wave * 16;                              // wave owns 16 output rows
  const float scale = 0.0883883476483184f;           // 1/sqrt(H=128)
  size_t off = (size_t)bc * HW_PLANE;

  v8f accO[8] = {};                                  // 16x128 f32 output tile
#pragma unroll 1
  for (int br = 0; br < 2; ++br) {
    const __bf16* Q  = (br ? Q2  : Q1)  + off;
    const __bf16* K  = (br ? K2  : K1)  + off;
    const __bf16* Vt = (br ? V2T : V1T) + off;

    // S = Q * K^T  (16x128 tile per wave, K-loop over w)
    v8f acc[8] = {};
#pragma unroll
    for (int kk = 0; kk < 4; ++kk) {
      int k0 = kk * 32;
      v16bf a = load_a_bf16(Q, h0, k0, W_DIM);
#pragma unroll
      for (int t = 0; t < 8; ++t) {
        v16bf bf = load_bT_bf16(K, t * 16, k0, W_DIM);  // contiguous K rows
        acc[t] = wmma_bf16(a, bf, acc[t]);
      }
    }
    // row-wise softmax over g (128 cols = 8 tiles x 16 lanes-per-half)
#pragma unroll
    for (int j = 0; j < 8; ++j) {
      float m = acc[0][j] * scale;
#pragma unroll
      for (int t = 1; t < 8; ++t) m = fmaxf(m, acc[t][j] * scale);
#pragma unroll
      for (int mask = 1; mask <= 8; mask <<= 1) m = fmaxf(m, __shfl_xor(m, mask, 32));
      float s = 0.f;
#pragma unroll
      for (int t = 0; t < 8; ++t) { float e = __expf(acc[t][j] * scale - m); acc[t][j] = e; s += e; }
#pragma unroll
      for (int mask = 1; mask <= 8; mask <<= 1) s += __shfl_xor(s, mask, 32);
      float inv = 1.f / s;
      int row = j + 8 * half;
#pragma unroll
      for (int t = 0; t < 8; ++t)
        lds_s[wave][row][t * 16 + (lane & 15)] = (__bf16)(acc[t][j] * inv);
    }
    __syncthreads();
    // Out += softmax(S) * V   (B-fragments from pre-transposed V: contiguous)
#pragma unroll
    for (int kk = 0; kk < 4; ++kk) {
      int k0 = kk * 32;
      v16bf a = load_a_bf16(&lds_s[wave][0][0], 0, k0, W_DIM);
#pragma unroll
      for (int t = 0; t < 8; ++t) {
        v16bf bf = load_bT_bf16(Vt, t * 16, k0, W_DIM);
        accO[t] = wmma_bf16(a, bf, accO[t]);
      }
    }
    __syncthreads();
  }
  // write fused = csi1 + csi2; accumulate per-channel sum / sumsq
  float lsum = 0.f, lss = 0.f;
#pragma unroll
  for (int t = 0; t < 8; ++t) {
#pragma unroll
    for (int j = 0; j < 8; ++j) {
      int h = h0 + j + 8 * half;
      int w = t * 16 + (lane & 15);
      float v = accO[t][j];
      fused[off + (size_t)h * W_DIM + w] = v;
      lsum += v; lss += v * v;
    }
  }
#pragma unroll
  for (int mask = 1; mask <= 16; mask <<= 1) {
    lsum += __shfl_xor(lsum, mask, 32);
    lss  += __shfl_xor(lss,  mask, 32);
  }
  if (lane == 0) { atomicAdd(&chSum[c], lsum); atomicAdd(&chSumSq[c], lss); }
}

// ---- Kernel 3: BatchNorm(axis 0,2,3) + ReLU, float4 vectorized -------------
__global__ __launch_bounds__(256) void RF_bn_relu(
    const float* __restrict__ fused, const float* __restrict__ chSum,
    const float* __restrict__ chSumSq, const float* __restrict__ gamma,
    const float* __restrict__ beta, float* __restrict__ out) {
  size_t i    = (size_t)blockIdx.x * blockDim.x + threadIdx.x;  // float4 index
  size_t base = i * 4;
  int c = (int)((base / HW_PLANE) % C_DIM);
  const float invN = 1.f / (float)((size_t)B_DIM * HW_PLANE);
  float mean = chSum[c] * invN;
  float var  = chSumSq[c] * invN - mean * mean;
  float rs   = rsqrtf(var + 1e-5f);
  float g    = gamma[c] * rs;
  float bb   = beta[c] - mean * g;
  float4 v = ((const float4*)fused)[i];
  float4 r;
  r.x = fmaxf(v.x * g + bb, 0.f);
  r.y = fmaxf(v.y * g + bb, 0.f);
  r.z = fmaxf(v.z * g + bb, 0.f);
  r.w = fmaxf(v.w * g + bb, 0.f);
  ((float4*)out)[i] = r;
}

extern "C" void kernel_launch(void* const* d_in, const int* in_sizes, int n_in,
                              void* d_out, int out_size, void* d_ws, size_t ws_size,
                              hipStream_t stream) {
  const float* x1    = (const float*)d_in[0];
  const float* x2    = (const float*)d_in[1];
  const float* rssi1 = (const float*)d_in[2];
  const float* rssi2 = (const float*)d_in[3];
  const float* qw    = (const float*)d_in[4];
  const float* qb    = (const float*)d_in[5];
  const float* kw    = (const float*)d_in[6];
  const float* kb    = (const float*)d_in[7];
  const float* vw    = (const float*)d_in[8];
  const float* vb    = (const float*)d_in[9];
  const float* gamma = (const float*)d_in[10];
  const float* beta  = (const float*)d_in[11];
  float* out = (float*)d_out;

  // Workspace: 6 bf16 tensors (Q/K/V^T x2 branches) + f32 fused + stats
  const size_t SZ = (size_t)N_TOTAL;
  __bf16* Q1  = (__bf16*)d_ws;
  __bf16* K1  = Q1  + SZ;
  __bf16* V1T = K1  + SZ;
  __bf16* Q2  = V1T + SZ;
  __bf16* K2  = Q2  + SZ;
  __bf16* V2T = K2  + SZ;
  float* fused   = (float*)(V2T + SZ);
  float* chSum   = fused + SZ;
  float* chSumSq = chSum + C_DIM;

  RF_init_stats<<<1, 128, 0, stream>>>(chSum);
  RF_proj<<<dim3(HW_PLANE / 128, B_DIM, 6), 256, 0, stream>>>(
      x1, x2, rssi1, rssi2, qw, qb, kw, kb, vw, vb, Q1, K1, V1T, Q2, K2, V2T);
  RF_attn<<<dim3(B_DIM * C_DIM), 256, 0, stream>>>(
      Q1, K1, V1T, Q2, K2, V2T, fused, chSum, chSumSq);
  RF_bn_relu<<<dim3(N_TOTAL / 4 / 256), 256, 0, stream>>>(
      fused, chSum, chSumSq, gamma, beta, out);
}
